// FIBERTransformerBlock_55843164783180
// MI455X (gfx1250) — compile-verified, matrix operand order checked by reading, not run on hardware
//
#include <hip/hip_runtime.h>
#include <math.h>

// ---------------- problem constants ----------------
#define Bc   16
#define SVc  576
#define STc  512
#define Hc   768
#define NHc  12
#define DHc  64
#define Ic   3072
#define NVc  (Bc * SVc)   // 9216 vision tokens
#define NTc  (Bc * STc)   // 8192 text tokens
#define BHc  (Bc * NHc)   // 192 (batch*heads)
#define SCALEc 0.125f     // 1/sqrt(64)

typedef __bf16 bf16;
typedef __attribute__((ext_vector_type(16))) __bf16 v16bf;
typedef __attribute__((ext_vector_type(8)))  float  v8f;

union Frag {
  v16bf v;
  uint4 q[2];
};

// ---------------------------------------------------------------------------
// Kernel 1: wave-tile batched GEMM (used for attention GEMMs, small N).
// C[z] = alpha * A[z](MxK) @ W[z]^T(NxK) [+bias][gelu][+res]; one wave = 32x64 tile.
// ---------------------------------------------------------------------------
template<bool HB, bool HR, bool GE>
__global__ void __launch_bounds__(32)
gemm_bf16_kernel(const bf16* __restrict__ A, const bf16* __restrict__ Bw,
                 const float* __restrict__ bias, const float* __restrict__ res,
                 float* __restrict__ C, int K,
                 long sA, long sB, long sCb, long sCh, int nhC, int ldc,
                 float alpha)
{
  const int z = blockIdx.z;
  A  += (long)z * sA;
  Bw += (long)z * sB;
  const long coff = (long)(z / nhC) * sCb + (long)(z % nhC) * sCh;
  C += coff;

  const int lane = threadIdx.x;
  const int g    = lane >> 4;
  const int r    = lane & 15;
  const int m0   = blockIdx.y * 32;
  const int n0   = blockIdx.x * 64;

  const bf16* arow[2];
  arow[0] = A + (long)(m0 + r)      * K;
  arow[1] = A + (long)(m0 + 16 + r) * K;
  const bf16* brow[4];
#pragma unroll
  for (int nt = 0; nt < 4; ++nt)
    brow[nt] = Bw + (long)(n0 + nt * 16 + r) * K;

  v8f acc[2][4];
#pragma unroll
  for (int mt = 0; mt < 2; ++mt)
#pragma unroll
    for (int nt = 0; nt < 4; ++nt)
#pragma unroll
      for (int e = 0; e < 8; ++e) acc[mt][nt][e] = 0.0f;

  const int kg = g * 8;
  for (int k = 0; k < K; k += 32) {
    Frag a[2], b[4];
#pragma unroll
    for (int mt = 0; mt < 2; ++mt) {
      const bf16* p = arow[mt] + k + kg;
      a[mt].q[0] = *reinterpret_cast<const uint4*>(p);
      a[mt].q[1] = *reinterpret_cast<const uint4*>(p + 16);
    }
#pragma unroll
    for (int nt = 0; nt < 4; ++nt) {
      const bf16* p = brow[nt] + k + kg;
      b[nt].q[0] = *reinterpret_cast<const uint4*>(p);
      b[nt].q[1] = *reinterpret_cast<const uint4*>(p + 16);
    }
#pragma unroll
    for (int mt = 0; mt < 2; ++mt)
#pragma unroll
      for (int nt = 0; nt < 4; ++nt)
        acc[mt][nt] = __builtin_amdgcn_wmma_f32_16x16x32_bf16(
            false, a[mt].v, false, b[nt].v, (short)0, acc[mt][nt], false, false);
  }

#pragma unroll
  for (int nt = 0; nt < 4; ++nt) {
    const int n  = n0 + nt * 16 + r;
    const float bv = HB ? bias[n] : 0.0f;
#pragma unroll
    for (int mt = 0; mt < 2; ++mt) {
      const int mbase = m0 + mt * 16 + 8 * g;
#pragma unroll
      for (int e = 0; e < 8; ++e) {
        const int m = mbase + e;
        float v = acc[mt][nt][e] * alpha;
        if (HB) v += bv;
        if (GE) v = 0.5f * v * (1.0f + erff(v * 0.70710678118654752f));
        const long ci = (long)m * ldc + n;
        if (HR) v += res[ci];
        C[ci] = v;
      }
    }
  }
}

// ---------------------------------------------------------------------------
// Kernel 2: LDS-staged GEMM via CDNA5 async-to-LDS path (projection / FFN GEMMs).
// 128 threads = 4 waves (2x2), block tile 64(M) x 128(N), K-step 32.
// A-tile 64x32 bf16 and B-tile 128x32 bf16 staged into LDS with
// global_load_async_to_lds_b128 (ASYNCcnt), double buffered.
// LDS rows padded to 80B so 16 consecutive rows hit disjoint bank groups.
// ---------------------------------------------------------------------------
#define ROWB    80
#define LDSA_SZ (64 * ROWB)             // 5120 B
#define LDSB_SZ (128 * ROWB)            // 10240 B
#define LDS_BUF (LDSA_SZ + LDSB_SZ)     // 15360 B

template<bool HB, bool HR, bool GE>
__global__ void __launch_bounds__(128)
gemm_bf16_lds_kernel(const bf16* __restrict__ A, const bf16* __restrict__ Bw,
                     const float* __restrict__ bias, const float* __restrict__ res,
                     float* __restrict__ C, int K, int ldc, float alpha)
{
  __shared__ uint4 ldsmem[(2 * LDS_BUF) / 16];
  char* ldsb = (char*)ldsmem;

  const int tid   = threadIdx.x;
  const int lane  = tid & 31;
  const int w     = tid >> 5;
  const int waveM = w & 1;
  const int waveN = w >> 1;
  const int g     = lane >> 4;
  const int r     = lane & 15;
  const int m0    = blockIdx.y * 64;
  const int n0    = blockIdx.x * 128;
  const long Kb   = (long)K * 2;         // row length in bytes

  // per-thread async chunk descriptors (16B chunks; 4 chunks per 64B K-slab row)
  unsigned voffA[2], ldsAo[2];
#pragma unroll
  for (int j = 0; j < 2; ++j) {
    const int cid = tid + j * 128;       // 0..255  -> A tile 64 rows x 4 chunks
    const int row = cid >> 2, kc = cid & 3;
    voffA[j] = (unsigned)((long)(m0 + row) * Kb + kc * 16);
    ldsAo[j] = row * ROWB + kc * 16;
  }
  unsigned voffB[4], ldsBo[4];
#pragma unroll
  for (int j = 0; j < 4; ++j) {
    const int cid = tid + j * 128;       // 0..511  -> B tile 128 rows x 4 chunks
    const int row = cid >> 2, kc = cid & 3;
    voffB[j] = (unsigned)((long)(n0 + row) * Kb + kc * 16);
    ldsBo[j] = LDSA_SZ + row * ROWB + kc * 16;
  }
  const unsigned long long abase = (unsigned long long)(size_t)A;
  const unsigned long long bbase = (unsigned long long)(size_t)Bw;
  // ISA 10.2: generic LDS pointer keeps the LDS byte offset in addr[31:0].
  const unsigned ldsbase = (unsigned)(size_t)ldsb;

  v8f acc[2][4];
#pragma unroll
  for (int mt = 0; mt < 2; ++mt)
#pragma unroll
    for (int nt = 0; nt < 4; ++nt)
#pragma unroll
      for (int e = 0; e < 8; ++e) acc[mt][nt][e] = 0.0f;

  auto issue = [&](int buf) {
    const unsigned bo = ldsbase + buf * LDS_BUF;
#pragma unroll
    for (int j = 0; j < 2; ++j)
      asm volatile("global_load_async_to_lds_b128 %0, %1, %2"
                   :: "v"(bo + ldsAo[j]), "v"(voffA[j]), "s"(abase) : "memory");
#pragma unroll
    for (int j = 0; j < 4; ++j)
      asm volatile("global_load_async_to_lds_b128 %0, %1, %2"
                   :: "v"(bo + ldsBo[j]), "v"(voffB[j]), "s"(bbase) : "memory");
#pragma unroll
    for (int j = 0; j < 2; ++j) voffA[j] += 64;
#pragma unroll
    for (int j = 0; j < 4; ++j) voffB[j] += 64;
  };

  auto compute = [&](int buf) {
    const char* aT = ldsb + buf * LDS_BUF;
    const char* bT = aT + LDSA_SZ;
    Frag a[2], b[4];
#pragma unroll
    for (int mt = 0; mt < 2; ++mt) {
      const char* p = aT + (waveM * 32 + mt * 16 + r) * ROWB + g * 16;
      a[mt].q[0] = *reinterpret_cast<const uint4*>(p);
      a[mt].q[1] = *reinterpret_cast<const uint4*>(p + 32);
    }
#pragma unroll
    for (int nt = 0; nt < 4; ++nt) {
      const char* p = bT + (waveN * 64 + nt * 16 + r) * ROWB + g * 16;
      b[nt].q[0] = *reinterpret_cast<const uint4*>(p);
      b[nt].q[1] = *reinterpret_cast<const uint4*>(p + 32);
    }
#pragma unroll
    for (int mt = 0; mt < 2; ++mt)
#pragma unroll
      for (int nt = 0; nt < 4; ++nt)
        acc[mt][nt] = __builtin_amdgcn_wmma_f32_16x16x32_bf16(
            false, a[mt].v, false, b[nt].v, (short)0, acc[mt][nt], false, false);
  };

  int buf = 0;
  issue(buf);
  asm volatile("s_wait_asynccnt 0x0" ::: "memory");
  __syncthreads();
  for (int k = 32; k < K; k += 32) {
    issue(buf ^ 1);           // prefetch next K-slab while computing current
    compute(buf);
    asm volatile("s_wait_asynccnt 0x0" ::: "memory");
    __syncthreads();
    buf ^= 1;
  }
  compute(buf);

  // epilogue
#pragma unroll
  for (int nt = 0; nt < 4; ++nt) {
    const int n  = n0 + waveN * 64 + nt * 16 + r;
    const float bv = HB ? bias[n] : 0.0f;
#pragma unroll
    for (int mt = 0; mt < 2; ++mt) {
      const int mbase = m0 + waveM * 32 + mt * 16 + 8 * g;
#pragma unroll
      for (int e = 0; e < 8; ++e) {
        const int m = mbase + e;
        float v = acc[mt][nt][e] * alpha;
        if (HB) v += bv;
        if (GE) v = 0.5f * v * (1.0f + erff(v * 0.70710678118654752f));
        const long ci = (long)m * ldc + n;
        if (HR) v += res[ci];
        C[ci] = v;
      }
    }
  }
}

// ---------------------------------------------------------------------------
__global__ void cvt_f32_bf16(const float* __restrict__ s, bf16* __restrict__ d, long n)
{
  long i = (long)blockIdx.x * blockDim.x + threadIdx.x;
  if (i < n) d[i] = (bf16)s[i];
}

__global__ void pack_heads(const float* __restrict__ src, bf16* __restrict__ dst,
                           int S, int rowLen, int colOff, long tot)
{
  long i = (long)blockIdx.x * blockDim.x + threadIdx.x;
  if (i >= tot) return;
  int  d  = (int)(i & (DHc - 1));
  long t  = i >> 6;
  int  s  = (int)(t % S);
  long bh = t / S;
  int  h  = (int)(bh % NHc);
  int  b  = (int)(bh / NHc);
  dst[i] = (bf16)src[((long)b * S + s) * rowLen + colOff + h * DHc + d];
}

__global__ void pack_headsT(const float* __restrict__ src, bf16* __restrict__ dst,
                            int S, int rowLen, int colOff, long tot)
{
  long i = (long)blockIdx.x * blockDim.x + threadIdx.x;
  if (i >= tot) return;
  int  s  = (int)(i % S);
  long t  = i / S;
  int  d  = (int)(t & (DHc - 1));
  long bh = t >> 6;
  int  h  = (int)(bh % NHc);
  int  b  = (int)(bh / NHc);
  dst[i] = (bf16)src[((long)b * S + s) * rowLen + colOff + h * DHc + d];
}

__global__ void softmax_bf16(const float* __restrict__ s, bf16* __restrict__ p, int len)
{
  __shared__ float red[256];
  const long base = (long)blockIdx.x * len;
  const int  tid  = threadIdx.x;

  float mx = -3.4e38f;
  for (int i = tid; i < len; i += 256) mx = fmaxf(mx, s[base + i]);
  red[tid] = mx; __syncthreads();
  for (int o = 128; o > 0; o >>= 1) {
    if (tid < o) red[tid] = fmaxf(red[tid], red[tid + o]);
    __syncthreads();
  }
  mx = red[0]; __syncthreads();

  float sum = 0.0f;
  for (int i = tid; i < len; i += 256) sum += expf(s[base + i] - mx);
  red[tid] = sum; __syncthreads();
  for (int o = 128; o > 0; o >>= 1) {
    if (tid < o) red[tid] += red[tid + o];
    __syncthreads();
  }
  const float inv = 1.0f / red[0];

  for (int i = tid; i < len; i += 256)
    p[base + i] = (bf16)(expf(s[base + i] - mx) * inv);
}

__global__ void layernorm_rows(const float* __restrict__ x, const float* __restrict__ g,
                               const float* __restrict__ b, float* __restrict__ y, int H)
{
  __shared__ float red[256];
  const long base = (long)blockIdx.x * H;
  const int  tid  = threadIdx.x;

  float sum = 0.0f;
  for (int i = tid; i < H; i += 256) sum += x[base + i];
  red[tid] = sum; __syncthreads();
  for (int o = 128; o > 0; o >>= 1) {
    if (tid < o) red[tid] += red[tid + o];
    __syncthreads();
  }
  const float mean = red[0] / (float)H; __syncthreads();

  float vs = 0.0f;
  for (int i = tid; i < H; i += 256) { float d = x[base + i] - mean; vs += d * d; }
  red[tid] = vs; __syncthreads();
  for (int o = 128; o > 0; o >>= 1) {
    if (tid < o) red[tid] += red[tid + o];
    __syncthreads();
  }
  const float inv = rsqrtf(red[0] / (float)H + 1e-12f);

  for (int i = tid; i < H; i += 256)
    y[base + i] = (x[base + i] - mean) * inv * g[i] + b[i];
}

// ---------------------------------------------------------------------------
template<bool HB, bool HR, bool GE>
static void gemm_launch(hipStream_t st, const bf16* A, const bf16* W,
                        const float* bias, const float* res, float* C,
                        int M, int N, int K,
                        long sA, long sB, long sCb, long sCh, int nhC, int ldc,
                        int batch, float alpha)
{
  dim3 grid(N / 64, M / 32, batch), blk(32, 1, 1);
  gemm_bf16_kernel<HB, HR, GE><<<grid, blk, 0, st>>>(
      A, W, bias, res, C, K, sA, sB, sCb, sCh, nhC, ldc, alpha);
}

template<bool HB, bool HR, bool GE>
static void gemm_lds_launch(hipStream_t st, const bf16* A, const bf16* W,
                            const float* bias, const float* res, float* C,
                            int M, int N, int K, float alpha)
{
  dim3 grid(N / 128, M / 64, 1), blk(128, 1, 1);
  gemm_bf16_lds_kernel<HB, HR, GE><<<grid, blk, 0, st>>>(
      A, W, bias, res, C, K, /*ldc=*/N, alpha);
}

// input indices (setup_inputs dict order)
enum {
  IN_V = 0, IN_T = 1,
  W_VA_IN = 2, W_VA_OUT = 3, W_TA_IN = 4, W_TA_OUT = 5,
  W_V2T_Q = 6, W_V2T_K = 7, W_V2T_V = 8,
  W_T2V_Q = 9, W_T2V_K = 10, W_T2V_V = 11,
  W_VOUT = 12, W_TOUT = 13,
  W_VINT = 14, W_VFF = 15, W_TINT = 16, W_TFF = 17,
  B_VA_IN = 18, B_VA_OUT = 19, B_TA_IN = 20, B_TA_OUT = 21,
  B_V2T_Q = 22, B_V2T_K = 23, B_V2T_V = 24,
  B_T2V_Q = 25, B_T2V_K = 26, B_T2V_V = 27,
  B_VOUT = 28, B_TOUT = 29,
  B_VINT = 30, B_VFF = 31, B_TINT = 32, B_TFF = 33,
  B_VLN = 34, B_TLN = 35, B_VLN1 = 36, B_VLN2 = 37, B_TLN1 = 38, B_TLN2 = 39,
  G_VLN = 40, G_TLN = 41, G_VLN1 = 42, G_VLN2 = 43, G_TLN1 = 44, G_TLN2 = 45,
};

extern "C" void kernel_launch(void* const* d_in, const int* in_sizes, int n_in,
                              void* d_out, int out_size, void* d_ws, size_t ws_size,
                              hipStream_t stream)
{
  (void)n_in; (void)out_size; (void)ws_size;
  char* wsp = (char*)d_ws;
  size_t off = 0;
  auto alloc = [&](size_t bytes) -> void* {
    void* p = wsp + off;
    off += (bytes + 255) & ~(size_t)255;
    return p;
  };
  auto F = [&](int i) { return (const float*)d_in[i]; };

  auto cvt = [&](const float* s, bf16* d, long n) {
    int blocks = (int)((n + 255) / 256);
    cvt_f32_bf16<<<blocks, 256, 0, stream>>>(s, d, n);
  };

  // ---- bf16 weights ----
  bf16* wbf[18] = {};
  for (int i = 2; i <= 17; ++i) {
    wbf[i] = (bf16*)alloc((size_t)in_sizes[i] * sizeof(bf16));
    cvt(F(i), wbf[i], in_sizes[i]);
  }

  // ---- activation buffers ----
  float* vbuf = (float*)alloc((size_t)NVc * Hc * 4);
  float* tbuf = (float*)alloc((size_t)NTc * Hc * 4);
  float* vtmp = (float*)alloc((size_t)NVc * Hc * 4);
  float* ttmp = (float*)alloc((size_t)NTc * Hc * 4);
  bf16*  vbfb = (bf16*)alloc((size_t)NVc * Hc * 2);
  bf16*  tbfb = (bf16*)alloc((size_t)NTc * Hc * 2);
  bf16*  Qv   = (bf16*)alloc((size_t)NVc * Hc * 2);
  bf16*  Kv   = (bf16*)alloc((size_t)NVc * Hc * 2);
  bf16*  VTv  = (bf16*)alloc((size_t)NVc * Hc * 2);
  bf16*  Qt   = (bf16*)alloc((size_t)NTc * Hc * 2);
  bf16*  Kt   = (bf16*)alloc((size_t)NTc * Hc * 2);
  bf16*  VTt  = (bf16*)alloc((size_t)NTc * Hc * 2);
  float* Sreg = (float*)alloc((size_t)BHc * SVc * SVc * 4); // scores / qkv / ffn fp32
  bf16*  Preg = (bf16*)alloc((size_t)BHc * SVc * SVc * 2);  // probs / ffn bf16

  float* qkv   = Sreg;
  float* ffn32 = Sreg;
  bf16*  ffnbf = Preg;

  auto packQ = [&](const float* src, bf16* dst, int S, int rowLen, int colOff, long tot) {
    pack_heads<<<(int)((tot + 255) / 256), 256, 0, stream>>>(src, dst, S, rowLen, colOff, tot);
  };
  auto packT = [&](const float* src, bf16* dst, int S, int rowLen, int colOff, long tot) {
    pack_headsT<<<(int)((tot + 255) / 256), 256, 0, stream>>>(src, dst, S, rowLen, colOff, tot);
  };

  // =====================  per-modality self-attention  =====================
  auto self_attn = [&](const float* x, int Ntok, int S,
                       const bf16* Win, const float* bin,
                       const bf16* Wout, const float* bout,
                       const float* lng, const float* lnb,
                       bf16* xbf, float* tmp, float* outb,
                       bf16* Qh, bf16* Kh, bf16* VT) {
    const long tot = (long)Ntok * Hc;
    cvt(x, xbf, tot);
    gemm_lds_launch<true, false, false>(stream, xbf, Win, bin, nullptr, qkv,
                                        Ntok, 3 * Hc, Hc, 1.0f);
    packQ(qkv, Qh, S, 3 * Hc, 0, tot);
    packQ(qkv, Kh, S, 3 * Hc, Hc, tot);
    packT(qkv, VT, S, 3 * Hc, 2 * Hc, tot);
    gemm_launch<false, false, false>(stream, Qh, Kh, nullptr, nullptr, Sreg,
                                     S, S, DHc, (long)S * DHc, (long)S * DHc,
                                     (long)S * S, 0, 1, S, BHc, SCALEc);
    softmax_bf16<<<BHc * S, 256, 0, stream>>>(Sreg, Preg, S);
    gemm_launch<false, false, false>(stream, Preg, VT, nullptr, nullptr, tmp,
                                     S, DHc, S, (long)S * S, (long)DHc * S,
                                     (long)S * Hc, DHc, NHc, Hc, BHc, 1.0f);
    cvt(tmp, xbf, tot);
    gemm_lds_launch<true, true, false>(stream, xbf, Wout, bout, x, tmp,
                                       Ntok, Hc, Hc, 1.0f);
    layernorm_rows<<<Ntok, 256, 0, stream>>>(tmp, lng, lnb, outb, Hc);
  };

  self_attn(F(IN_V), NVc, SVc, wbf[W_VA_IN], F(B_VA_IN), wbf[W_VA_OUT], F(B_VA_OUT),
            F(G_VLN1), F(B_VLN1), vbfb, vtmp, vbuf, Qv, Kv, VTv);
  self_attn(F(IN_T), NTc, STc, wbf[W_TA_IN], F(B_TA_IN), wbf[W_TA_OUT], F(B_TA_OUT),
            F(G_TLN1), F(B_TLN1), tbfb, ttmp, tbuf, Qt, Kt, VTt);

  // =====================  cross-modal attention  =====================
  cvt(vbuf, vbfb, (long)NVc * Hc);
  cvt(tbuf, tbfb, (long)NTc * Hc);

  auto proj_heads = [&](const bf16* xbf, int Ntok, int S, const bf16* W,
                        const float* bias, bf16* dst, bool transposed) {
    gemm_lds_launch<true, false, false>(stream, xbf, W, bias, nullptr, vtmp,
                                        Ntok, Hc, Hc, 1.0f);
    const long tot = (long)Ntok * Hc;
    if (transposed) packT(vtmp, dst, S, Hc, 0, tot);
    else            packQ(vtmp, dst, S, Hc, 0, tot);
  };

  proj_heads(vbfb, NVc, SVc, wbf[W_V2T_Q], F(B_V2T_Q), Qv,  false);
  proj_heads(vbfb, NVc, SVc, wbf[W_V2T_K], F(B_V2T_K), Kv,  false);
  proj_heads(vbfb, NVc, SVc, wbf[W_V2T_V], F(B_V2T_V), VTv, true);
  proj_heads(tbfb, NTc, STc, wbf[W_T2V_Q], F(B_T2V_Q), Qt,  false);
  proj_heads(tbfb, NTc, STc, wbf[W_T2V_K], F(B_T2V_K), Kt,  false);
  proj_heads(tbfb, NTc, STc, wbf[W_T2V_V], F(B_T2V_V), VTt, true);

  // v2t: vision queries attend to text keys/values
  gemm_launch<false, false, false>(stream, Qv, Kt, nullptr, nullptr, Sreg,
                                   SVc, STc, DHc, (long)SVc * DHc, (long)STc * DHc,
                                   (long)SVc * STc, 0, 1, STc, BHc, SCALEc);
  softmax_bf16<<<BHc * SVc, 256, 0, stream>>>(Sreg, Preg, STc);
  gemm_launch<false, false, false>(stream, Preg, VTt, nullptr, nullptr, vtmp,
                                   SVc, DHc, STc, (long)SVc * STc, (long)DHc * STc,
                                   (long)SVc * Hc, DHc, NHc, Hc, BHc, 1.0f);

  // t2v: text queries attend to vision keys/values
  gemm_launch<false, false, false>(stream, Qt, Kv, nullptr, nullptr, Sreg,
                                   STc, SVc, DHc, (long)STc * DHc, (long)SVc * DHc,
                                   (long)STc * SVc, 0, 1, SVc, BHc, SCALEc);
  softmax_bf16<<<BHc * STc, 256, 0, stream>>>(Sreg, Preg, SVc);
  gemm_launch<false, false, false>(stream, Preg, VTv, nullptr, nullptr, ttmp,
                                   STc, DHc, SVc, (long)STc * SVc, (long)DHc * SVc,
                                   (long)STc * Hc, DHc, NHc, Hc, BHc, 1.0f);

  // cross out-proj + residual + LN
  cvt(vtmp, vbfb, (long)NVc * Hc);
  gemm_lds_launch<true, true, false>(stream, vbfb, wbf[W_VOUT], F(B_VOUT), vbuf, vtmp,
                                     NVc, Hc, Hc, 1.0f);
  layernorm_rows<<<NVc, 256, 0, stream>>>(vtmp, F(G_VLN), F(B_VLN), vbuf, Hc);

  cvt(ttmp, tbfb, (long)NTc * Hc);
  gemm_lds_launch<true, true, false>(stream, tbfb, wbf[W_TOUT], F(B_TOUT), tbuf, ttmp,
                                     NTc, Hc, Hc, 1.0f);
  layernorm_rows<<<NTc, 256, 0, stream>>>(ttmp, F(G_TLN), F(B_TLN), tbuf, Hc);

  // =====================  FFN + residual + LN2 -> outputs  =====================
  float* out_v = (float*)d_out;
  float* out_t = (float*)d_out + (long)NVc * Hc;

  auto ffn = [&](float* xstate, int Ntok,
                 const bf16* Wint, const float* bint,
                 const bf16* Wff, const float* bff,
                 const float* lng, const float* lnb,
                 bf16* xbf, float* tmp, float* outp) {
    cvt(xstate, xbf, (long)Ntok * Hc);
    gemm_lds_launch<true, false, true>(stream, xbf, Wint, bint, nullptr, ffn32,
                                       Ntok, Ic, Hc, 1.0f);            // GELU fused
    cvt(ffn32, ffnbf, (long)Ntok * Ic);
    gemm_lds_launch<true, true, false>(stream, ffnbf, Wff, bff, xstate, tmp,
                                       Ntok, Hc, Ic, 1.0f);
    layernorm_rows<<<Ntok, 256, 0, stream>>>(tmp, lng, lnb, outp, Hc);
  };

  ffn(vbuf, NVc, wbf[W_VINT], F(B_VINT), wbf[W_VFF], F(B_VFF),
      F(G_VLN2), F(B_VLN2), vbfb, vtmp, out_v);
  ffn(tbuf, NTc, wbf[W_TINT], F(B_TINT), wbf[W_TFF], F(B_TFF),
      F(G_TLN2), F(B_TLN2), tbfb, ttmp, out_t);
}
